// getBoxInfoListForOneImage_23828478558288
// MI455X (gfx1250) — 8-bit matrix operand order repaired and checked by reading, compiled, not run on hardware
//
#include <hip/hip_runtime.h>
#include <hip/hip_bf16.h>
#include <math.h>

typedef __attribute__((ext_vector_type(8))) int v8i;

#define IMG_W   1536
#define STRIDE  2
#define W_OUT   768
#define H_OUT   768
#define N_BOX   64
#define TILES_X 48
#define TILES_Y 48
#define N_TILES (TILES_X * TILES_Y)
#define WAVES_PER_BLOCK 8

__global__ void build_masks_kernel(const float* __restrict__ boxes,
                                   unsigned long long* __restrict__ maskx,
                                   unsigned long long* __restrict__ masky,
                                   float* __restrict__ scores,
                                   int* __restrict__ valid)
{
    int i = blockIdx.x * blockDim.x + threadIdx.x;
    if (i < N_BOX) { scores[i] = 0.0f; valid[i] = 0; }
    if (i >= W_OUT) return;

    float c = (float)i * (float)STRIDE + (float)STRIDE * 0.5f;
    unsigned long long mx = 0ull, my = 0ull;
    for (int b = 0; b < N_BOX; ++b) {
        float x1 = boxes[b * 5 + 0];
        float y1 = boxes[b * 5 + 1];
        float x2 = boxes[b * 5 + 2];
        float y2 = boxes[b * 5 + 3];
        bool area_ok = ((x2 - x1) * (y2 - y1)) != 0.0f;
        if (c >= x1 && c <= x2)            mx |= (1ull << b);
        if (area_ok && c >= y1 && c <= y2) my |= (1ull << b);
    }
    maskx[i] = mx;
    masky[i] = my;
}

__global__ void __launch_bounds__(256)
tile_wmma_kernel(const float* __restrict__ conf,
                 const unsigned long long* __restrict__ maskx,
                 const unsigned long long* __restrict__ masky,
                 float* __restrict__ scores,
                 int* __restrict__ valid)
{
    __shared__ float lscore[N_BOX];
    __shared__ int   lvalid[N_BOX];

    const int tid = threadIdx.x;
    if (tid < N_BOX) { lscore[tid] = 0.0f; lvalid[tid] = 0; }
    __syncthreads();

    const int wave = tid >> 5;
    const int lane = tid & 31;
    const int hi   = lane >> 4;
    const int mn   = lane & 15;

    const int tileIdx = blockIdx.x * WAVES_PER_BLOCK + wave;
    const int y0 = (tileIdx / TILES_X) * 16;
    const int x0 = (tileIdx % TILES_X) * 16;

    const unsigned long long my = masky[y0 + mn];
    const unsigned long long mx = maskx[x0 + mn];

    v8i A  = {};
    v8i A2 = {};
    v8i B  = {};
#pragma unroll
    for (int v = 0; v < 8; ++v) {
        const int ka = ((v >> 1) << 4) + ((v & 1) << 2) + hi * 8;
        unsigned nib     = (unsigned)(my >> ka) & 0xFu;
        unsigned bytes01 = (nib * 0x00204081u) & 0x01010101u;
        A[v] = (int)bytes01;
        unsigned expand = bytes01 * 0xFFu;
        A2[v] = (int)(bytes01 * (unsigned)ka + (expand & 0x03020100u));

        const int kb = ((v >> 1) << 4) + ((v & 1) << 2) + hi * 8;
        unsigned nibb = (unsigned)(mx >> kb) & 0xFu;
        B[v] = (int)((nibb * 0x00204081u) & 0x01010101u);
    }

    const v8i zero = {};
    v8i cnt = __builtin_amdgcn_wmma_i32_16x16x64_iu8(false, A,  false, B, zero, false, false);
    v8i ids = __builtin_amdgcn_wmma_i32_16x16x64_iu8(false, A2, false, B, zero, false, false);

#pragma unroll
    for (int r = 0; r < 8; ++r) {
        if (cnt[r] == 1) {
            const int M = r + hi * 8;
            const float c = conf[(y0 + M) * W_OUT + (x0 + mn)];
            const float s = __builtin_amdgcn_rcpf(1.0f + __expf(-c));
            const int b = ids[r];
            atomicMax(&lscore[b], s);
            lvalid[b] = 1;
        }
    }
    __syncthreads();

    if (tid < N_BOX && lvalid[tid]) {
        atomicMax(&scores[tid], lscore[tid]);
        atomicOr(&valid[tid], 1);
    }
}

__global__ void finalize_kernel(const float* __restrict__ scores,
                                const int* __restrict__ valid,
                                float* __restrict__ out)
{
    int i = threadIdx.x;
    if (i < N_BOX) {
        out[i]         = scores[i];
        out[N_BOX + i] = valid[i] ? 1.0f : 0.0f;
    }
}

extern "C" void kernel_launch(void* const* d_in, const int* in_sizes, int n_in,
                              void* d_out, int out_size, void* d_ws, size_t ws_size,
                              hipStream_t stream)
{
    const float* conf  = (const float*)d_in[0];
    const float* boxes = (const float*)d_in[2];

    char* ws = (char*)d_ws;
    unsigned long long* masky  = (unsigned long long*)(ws);
    unsigned long long* maskx  = (unsigned long long*)(ws + H_OUT * sizeof(unsigned long long));
    float*              scores = (float*)(ws + 2 * H_OUT * sizeof(unsigned long long));
    int*                valid  = (int*)  (ws + 2 * H_OUT * sizeof(unsigned long long)
                                             + N_BOX * sizeof(float));

    build_masks_kernel<<<(W_OUT + 255) / 256, 256, 0, stream>>>(boxes, maskx, masky,
                                                                scores, valid);
    tile_wmma_kernel<<<N_TILES / WAVES_PER_BLOCK, 256, 0, stream>>>(conf, maskx, masky,
                                                                    scores, valid);
    finalize_kernel<<<1, 64, 0, stream>>>(scores, valid, (float*)d_out);
}
